// GraphEncoder_71657234366492
// MI455X (gfx1250) — matrix operand order repaired and checked by reading, not compiled, not run on hardware
//
#include <hip/hip_runtime.h>
#include <cstdint>

typedef __attribute__((ext_vector_type(16))) _Float16 v16h;
typedef __attribute__((ext_vector_type(8)))  float    v8f;

#define N_NODES 50000
#define N_EDGES 800000
#define HID     128
#define LAYERS  3

// ---------------- helpers ----------------

__global__ void zero_f32_kernel(float* __restrict__ p, int n) {
    int i = blockIdx.x * blockDim.x + threadIdx.x;
    if (i < n) p[i] = 0.0f;
}

__global__ void deg_kernel(const long long* __restrict__ dst,
                           float* __restrict__ deg, int E) {
    int e = blockIdx.x * blockDim.x + threadIdx.x;
    if (e < E) atomicAdd(&deg[(int)dst[e]], 1.0f);
}

// deg -> dinv in place
__global__ void dinv_kernel(float* __restrict__ deg, int n) {
    int i = blockIdx.x * blockDim.x + threadIdx.x;
    if (i < n) {
        float d = deg[i];
        deg[i] = (d > 0.0f) ? rsqrtf(fmaxf(d, 1.0f)) : 0.0f;
    }
}

__global__ void norm_kernel(const long long* __restrict__ src,
                            const long long* __restrict__ dst,
                            const float* __restrict__ dinv,
                            float* __restrict__ norm, int E) {
    int e = blockIdx.x * blockDim.x + threadIdx.x;
    if (e < E) norm[e] = dinv[(int)src[e]] * dinv[(int)dst[e]];
}

// Pre-pack W[l] (row-major K x N, f32) into the WMMA B-operand register
// layout (f16): for column tile nt and K-chunk kc, lane L holds 16 contiguous
// halves = B[K = kc*32 + (L>=16?16:0) + s][N = nt*16 + (L&15)], s = 0..15.
// Packed index: ((((l*8 + nt)*4 + kc)*32 + L)*16 + s.
__global__ void prepack_w_kernel(const float* __restrict__ W,
                                 _Float16* __restrict__ Wp, int total) {
    int idx = blockIdx.x * blockDim.x + threadIdx.x;
    if (idx >= total) return;
    int s    = idx & 15;
    int lane = (idx >> 4) & 31;
    int kc   = (idx >> 9) & 3;
    int nt   = (idx >> 11) & 7;
    int l    = idx >> 14;
    int k = kc * 32 + ((lane & 16) ? 16 : 0) + s;
    int n = nt * 16 + (lane & 15);
    Wp[idx] = (_Float16)W[(size_t)l * HID * HID + (size_t)k * HID + n];
}

// Y[n][d] = b[d] : bias pre-init so scatter atomics accumulate on top.
__global__ void bias_init_kernel(const float* __restrict__ b,
                                 float* __restrict__ Y, int total) {
    int i = blockIdx.x * blockDim.x + threadIdx.x;
    if (i < total) Y[i] = b[i & (HID - 1)];
}

__global__ void copy4_kernel(const float4* __restrict__ s,
                             float4* __restrict__ d, int n4) {
    int i = blockIdx.x * blockDim.x + threadIdx.x;
    if (i < n4) d[i] = s[i];
}

// ---------------- WMMA GEMM: H[N,128] = X[N,128] @ W[128,128] ----------------
// 160-thread blocks = 5 wave32 waves; wave w of block b owns the 16-row strip
// starting at (b*5+w)*16 (50000 = 625*5*16, exact). Each wave keeps its whole
// 16x128 A strip in registers (4 x v16h) and sweeps the 8 column tiles with
// pre-packed B operands (2 x global_load_b128 each). 32 WMMAs per wave.
__global__ __launch_bounds__(160) void gemm_wmma_f16_kernel(
    const float* __restrict__ X, const _Float16* __restrict__ Wp,
    float* __restrict__ H) {
    const int wave = threadIdx.x >> 5;
    const int lane = threadIdx.x & 31;
    const int m    = lane & 15;          // A row within strip / B,D column
    const bool hi  = lane >= 16;

    const int row0 = (blockIdx.x * 5 + wave) * 16;
    const float4* xr = (const float4*)(X + (size_t)(row0 + m) * HID);

    // A tiles (16x32 f16 each): lane<16 row M=m holds K = k0+{0..15};
    // lanes 16..31 hold K = k0+{16..31}.
    v16h a[4];
    #pragma unroll
    for (int kc = 0; kc < 4; ++kc) {
        const int ka = kc * 32 + (hi ? 16 : 0);
        float4 p0 = xr[(ka     ) >> 2];
        float4 p1 = xr[(ka +  4) >> 2];
        float4 p2 = xr[(ka +  8) >> 2];
        float4 p3 = xr[(ka + 12) >> 2];
        v16h t;
        t[0]  = (_Float16)p0.x; t[1]  = (_Float16)p0.y;
        t[2]  = (_Float16)p0.z; t[3]  = (_Float16)p0.w;
        t[4]  = (_Float16)p1.x; t[5]  = (_Float16)p1.y;
        t[6]  = (_Float16)p1.z; t[7]  = (_Float16)p1.w;
        t[8]  = (_Float16)p2.x; t[9]  = (_Float16)p2.y;
        t[10] = (_Float16)p2.z; t[11] = (_Float16)p2.w;
        t[12] = (_Float16)p3.x; t[13] = (_Float16)p3.y;
        t[14] = (_Float16)p3.z; t[15] = (_Float16)p3.w;
        a[kc] = t;
    }

    // Pre-packed B: lane's 16 halves are contiguous (32B -> 2 x b128 loads).
    const v16h* wlane = (const v16h*)Wp + lane;

    #pragma unroll
    for (int nt = 0; nt < 8; ++nt) {
        v8f acc = {};
        #pragma unroll
        for (int kc = 0; kc < 4; ++kc) {
            v16h b = wlane[(nt * 4 + kc) * 32];
            acc = __builtin_amdgcn_wmma_f32_16x16x32_f16(
                /*neg_a=*/false, a[kc], /*neg_b=*/false, b,
                /*c_mod=*/(short)0, acc, /*reuse_a=*/false, /*reuse_b=*/false);
        }
        // D layout: VGPR v -> M = (hi?8:0)+v, N = m.
        const int nout = nt * 16 + m;
        #pragma unroll
        for (int v = 0; v < 8; ++v)
            H[(size_t)(row0 + (hi ? 8 : 0) + v) * HID + nout] = acc[v];
    }
}

// ---------------- edge scatter: Y[dst] += H[src] * norm ----------------
// 32 threads per edge, each thread handles a float4 (128 feats = 32 lanes x 4).
__global__ void scatter_kernel(const float* __restrict__ H,
                               const long long* __restrict__ src,
                               const long long* __restrict__ dst,
                               const float* __restrict__ norm,
                               float* __restrict__ Y, int E) {
    long long t = (long long)blockIdx.x * blockDim.x + threadIdx.x;
    int e = (int)(t >> 5);
    if (e >= E) return;
    int c = ((int)t & 31) * 4;

    size_t s = (size_t)(int)src[e];
    size_t d = (size_t)(int)dst[e];
    float nm = norm[e];

    float4 h4 = *(const float4*)(H + s * HID + c);
    float* o = Y + d * HID + c;
    atomicAdd(o + 0, h4.x * nm);
    atomicAdd(o + 1, h4.y * nm);
    atomicAdd(o + 2, h4.z * nm);
    atomicAdd(o + 3, h4.w * nm);
}

// ---------------- launch ----------------

extern "C" void kernel_launch(void* const* d_in, const int* in_sizes, int n_in,
                              void* d_out, int out_size, void* d_ws, size_t ws_size,
                              hipStream_t stream) {
    const float*     x   = (const float*)d_in[0];
    const long long* ei  = (const long long*)d_in[1];   // int64 [2, E]
    const long long* src = ei;
    const long long* dst = ei + N_EDGES;
    const float*     Ws  = (const float*)d_in[2];       // [3,128,128]
    const float*     bs  = (const float*)d_in[3];       // [3,128]
    float*           out = (float*)d_out;               // 5 x [N,128]

    const size_t NF = (size_t)N_NODES * HID;

    // workspace layout (256B-aligned offsets)
    char* ws = (char*)d_ws;
    float*     deg = (float*)(ws + 0);               // 200,000 B -> pad 200,192
    float*     nrm = (float*)(ws + 200192);          // 3,200,000 B
    _Float16*  Wp  = (_Float16*)(ws + 3400192);      // 98,304 B (packed f16 W)
    float*     h   = (float*)(ws + 3498496);         // 25,600,000 B
    (void)ws_size; (void)in_sizes; (void)n_in; (void)out_size;

    const int TB = 256;

    // degrees -> dinv -> per-edge norm (recomputed every call: deterministic)
    zero_f32_kernel<<<(N_NODES + TB - 1) / TB, TB, 0, stream>>>(deg, N_NODES);
    deg_kernel<<<(N_EDGES + TB - 1) / TB, TB, 0, stream>>>(dst, deg, N_EDGES);
    dinv_kernel<<<(N_NODES + TB - 1) / TB, TB, 0, stream>>>(deg, N_NODES);
    norm_kernel<<<(N_EDGES + TB - 1) / TB, TB, 0, stream>>>(src, dst, deg, nrm, N_EDGES);

    // pack weights into WMMA B layout (f16) once per call
    prepack_w_kernel<<<(LAYERS * HID * HID + TB - 1) / TB, TB, 0, stream>>>(
        Ws, Wp, LAYERS * HID * HID);

    // embeddings[0] = input x  -> slice 1
    copy4_kernel<<<((int)(NF / 4) + TB - 1) / TB, TB, 0, stream>>>(
        (const float4*)x, (float4*)(out + NF), (int)(NF / 4));

    for (int l = 0; l < LAYERS; ++l) {
        const float* xin = (l == 0) ? x : (out + (size_t)(1 + l) * NF);
        float*       y   = out + (size_t)(2 + l) * NF;

        // h = xin @ W[l] : 625 blocks x 5 waves, one 16-row strip per wave
        gemm_wmma_f16_kernel<<<N_NODES / (16 * 5), 160, 0, stream>>>(
            xin, Wp + (size_t)l * HID * HID, h);

        // y = b[l] broadcast, then scatter-add normalized messages
        bias_init_kernel<<<((int)NF + TB - 1) / TB, TB, 0, stream>>>(
            bs + (size_t)l * HID, y, (int)NF);

        long long sthreads = (long long)N_EDGES * 32;
        scatter_kernel<<<(int)((sthreads + TB - 1) / TB), TB, 0, stream>>>(
            h, src, dst, nrm, y, N_EDGES);
    }

    // final output = last embedding -> slice 0
    copy4_kernel<<<((int)(NF / 4) + TB - 1) / TB, TB, 0, stream>>>(
        (const float4*)(out + 4 * NF), (float4*)out, (int)(NF / 4));
}
